// HC2STARModel_86955907875152
// MI455X (gfx1250) — matrix operand order repaired and checked by reading, not compiled, not run on hardware
//
#include <hip/hip_runtime.h>
#include <math.h>

// ---------------------------------------------------------------------------
// Types for gfx1250 WMMA (wave32): V_WMMA_F32_16X16X32_F16
// ---------------------------------------------------------------------------
typedef _Float16 v16h __attribute__((ext_vector_type(16)));
typedef _Float16 v8h  __attribute__((ext_vector_type(8)));
typedef float    v8f  __attribute__((ext_vector_type(8)));

union V16 { v16h v; v8h h[2]; };

// ---------------------------------------------------------------------------
// Problem constants
// ---------------------------------------------------------------------------
#define BATCH   16384
#define DIN     1024
#define H1DIM   512
#define H2DIM   256
#define H3DIM   128
#define NDOM    4
#define MROWS   32          // rows per workgroup (2 M-tiles of 16)

// workspace layout (in _Float16 elements)
#define WS_NORM   0ul
#define WS_CW1T   (WS_NORM + (size_t)BATCH*DIN)
#define WS_CW2T   (WS_CW1T + (size_t)H1DIM*DIN)
#define WS_CW3T   (WS_CW2T + (size_t)H2DIM*H1DIM)
#define WS_DW1T   (WS_CW3T + (size_t)H3DIM*H2DIM)
#define WS_DW2T   (WS_DW1T + (size_t)NDOM*H1DIM*DIN)
#define WS_DW3T   (WS_DW2T + (size_t)NDOM*H2DIM*H1DIM)

// ---------------------------------------------------------------------------
// Kernel A: transpose (D,K,N) f32 -> (D,N,K) f16
// ---------------------------------------------------------------------------
__global__ void transpose_to_f16(const float* __restrict__ src,
                                 _Float16* __restrict__ dst,
                                 int K, int N, int total) {
  int i = blockIdx.x * 256 + threadIdx.x;
  if (i >= total) return;
  int n = i % N;
  int r = i / N;
  int k = r % K;
  int d = r / K;
  dst[((size_t)d * N + n) * K + k] = (_Float16)src[i];
}

// ---------------------------------------------------------------------------
// Kernel B: LayerNorm + domain-conditional affine -> f16 (one wave per row)
// ---------------------------------------------------------------------------
__global__ __launch_bounds__(256) void ln_affine_f16(
    const float* __restrict__ x, const int* __restrict__ dom,
    const float* __restrict__ pn_w, const float* __restrict__ pn_b,
    _Float16* __restrict__ nrm) {
  const int row  = blockIdx.x * 8 + (threadIdx.x >> 5);
  const int lane = threadIdx.x & 31;
  const float* xr = x + (size_t)row * DIN;

  float s = 0.0f, s2 = 0.0f;
  for (int c = lane; c < DIN; c += 32) {
    float v = xr[c];
    s += v; s2 += v * v;
  }
#pragma unroll
  for (int m = 16; m >= 1; m >>= 1) {
    s  += __shfl_xor(s,  m, 32);
    s2 += __shfl_xor(s2, m, 32);
  }
  const float mean = s  * (1.0f / DIN);
  const float var  = s2 * (1.0f / DIN) - mean * mean;
  const float inv  = rsqrtf(var + 1e-5f);

  const int d = dom[row];
  const float* w = pn_w + (size_t)d * DIN;
  const float* b = pn_b + (size_t)d * DIN;
  _Float16* nr = nrm + (size_t)row * DIN;
  for (int c = lane; c < DIN; c += 32)
    nr[c] = (_Float16)(((xr[c] - mean) * inv) * w[c] + b[c]);
}

// ---------------------------------------------------------------------------
// WMMA building blocks (gfx1250, wave32)
// A fragment (16x32 f16): lane<16 holds row (lane), K=k0+0..15;
// lane>=16 holds the same row, K offset +16.
// B fragment (32x16 f16) from transposed (N,K) weights: lane holds column
// N = n0+(lane&15), K = k0 + 16*(lane>=16) .. +15 (contiguous 32B load).
// C/D: element r <-> M = r + 8*(lane>=16), N = n0+(lane&15).
// ---------------------------------------------------------------------------
template<int K, int MT, int NT, bool PF>
__device__ __forceinline__ void gemm_mn(const _Float16* Abase,
                                        const _Float16* __restrict__ Bt,
                                        int n0base, int lane,
                                        v8f (&acc)[MT][NT]) {
  const int hi  = (lane >> 4) & 1;
  const int kA  = hi * 16;
  const int kB  = hi * 16;
  const int nlo = lane & 15;
  v8f zero = {};
#pragma unroll
  for (int mt = 0; mt < MT; ++mt)
#pragma unroll
    for (int t = 0; t < NT; ++t) acc[mt][t] = zero;

  for (int k0 = 0; k0 < K; k0 += 32) {
    V16 a[MT];
#pragma unroll
    for (int mt = 0; mt < MT; ++mt) {
      const _Float16* ap = Abase + (size_t)(mt * 16 + nlo) * K + k0 + kA;
      a[mt].h[0] = *(const v8h*)(ap);
      a[mt].h[1] = *(const v8h*)(ap + 8);
    }
#pragma unroll
    for (int t = 0; t < NT; ++t) {
      const _Float16* bp = Bt + (size_t)(n0base + t * 16 + nlo) * K + (k0 + kB);
      if (PF && (k0 + 256 < K))   // 512B-ahead stream prefetch, WGP scope
        __builtin_prefetch((const void*)(bp + 256), 0, 3);
      v16h b = *(const v16h*)bp;
#pragma unroll
      for (int mt = 0; mt < MT; ++mt)
        acc[mt][t] = __builtin_amdgcn_wmma_f32_16x16x32_f16(false, a[mt].v, false, b,
                                                            (short)0, acc[mt][t],
                                                            false, false);
    }
  }
}

template<int N>
__device__ __forceinline__ void store_act(const v8f& acc, int mrow, int n0, int lane,
                                          const float* __restrict__ bias,
                                          _Float16* dst, bool doRelu) {
  const int hi8 = ((lane >> 4) & 1) * 8;
  const int n = n0 + (lane & 15);
  const float bv = bias[n];
#pragma unroll
  for (int r = 0; r < 8; ++r) {
    float v = acc[r] + bv;
    if (doRelu) v = fmaxf(v, 0.0f);
    dst[(mrow + r + hi8) * N + n] = (_Float16)v;
  }
}

// ---------------------------------------------------------------------------
// Kernel C: fused GEMM chain, 32 rows per workgroup, 8 wave32s.
// Groups 0..3 = domain nets (scatter selected rows into sSel);
// group 4 = center net, fused in place: sSel = center * tanh(sSel).
// ---------------------------------------------------------------------------
struct Params {
  const _Float16* nrm;
  const _Float16* cW1t; const _Float16* cW2t; const _Float16* cW3t;
  const _Float16* dW1t; const _Float16* dW2t; const _Float16* dW3t;
  const int* dom;
  const float *cb1, *cb2, *cb3, *db1, *db2, *db3;
  const float *fW1, *fb1, *fW2, *fb2, *dom_emb, *aW1, *ab1, *aW2, *ab2;
  float* out;
};

__global__ __launch_bounds__(256) void fused_star(Params P) {
  __shared__ __align__(16) unsigned char smem[65536];
  _Float16* sH1  = (_Float16*)smem;            // 32 x 512 f16 = 32768 B
  _Float16* sH2  = (_Float16*)(smem + 32768);  // 32 x 256 f16 = 16384 B
  float*    sSel = (float*)   (smem + 49152);  // 32 x 128 f32 = 16384 B

  const int row0 = blockIdx.x * MROWS;
  const int lane = threadIdx.x & 31;
  const int wave = threadIdx.x >> 5;
  const int nlo  = lane & 15;
  const int hi8  = ((lane >> 4) & 1) * 8;

  // pack the domain ids (2 bits each) of the 16 rows this lane owns in the
  // C-tile layout: slot (mt*8 + r) <-> row m = mt*16 + r + hi8
  unsigned dompack = 0;
#pragma unroll
  for (int mt = 0; mt < 2; ++mt)
#pragma unroll
    for (int r = 0; r < 8; ++r)
      dompack |= ((unsigned)P.dom[row0 + mt * 16 + r + hi8] & 3u) << ((mt * 8 + r) * 2);

  const _Float16* Abase1 = P.nrm + (size_t)row0 * DIN;

  for (int gi = 0; gi < 5; ++gi) {
    const bool isCenter = (gi == 4);
    const _Float16* W1 = isCenter ? P.cW1t : P.dW1t + (size_t)gi * H1DIM * DIN;
    const _Float16* W2 = isCenter ? P.cW2t : P.dW2t + (size_t)gi * H2DIM * H1DIM;
    const _Float16* W3 = isCenter ? P.cW3t : P.dW3t + (size_t)gi * H3DIM * H2DIM;
    const float* b1 = isCenter ? P.cb1 : P.db1 + gi * H1DIM;
    const float* b2 = isCenter ? P.cb2 : P.db2 + gi * H2DIM;
    const float* b3 = isCenter ? P.cb3 : P.db3 + gi * H3DIM;

    // ---- layer 1: 1024 -> 512 (wave: 2 M-tiles x 4 N-tiles) ----
    {
      v8f acc[2][4];
      gemm_mn<DIN, 2, 4, true>(Abase1, W1, wave * 64, lane, acc);
#pragma unroll
      for (int mt = 0; mt < 2; ++mt)
#pragma unroll
        for (int t = 0; t < 4; ++t)
          store_act<H1DIM>(acc[mt][t], mt * 16, wave * 64 + t * 16, lane, b1, sH1, true);
    }
    __syncthreads();

    // ---- layer 2: 512 -> 256 (A from LDS; wave: 2 x 2 tiles) ----
    {
      v8f acc[2][2];
      gemm_mn<H1DIM, 2, 2, true>(sH1, W2, wave * 32, lane, acc);
#pragma unroll
      for (int mt = 0; mt < 2; ++mt)
#pragma unroll
        for (int t = 0; t < 2; ++t)
          store_act<H2DIM>(acc[mt][t], mt * 16, wave * 32 + t * 16, lane, b2, sH2, true);
    }
    __syncthreads();

    // ---- layer 3: 256 -> 128, no relu (wave: 2 x 1 tiles) ----
    {
      v8f acc[2][1];
      gemm_mn<H2DIM, 2, 1, false>(sH2, W3, wave * 16, lane, acc);
      const int n = wave * 16 + nlo;
      const float bv = b3[n];
#pragma unroll
      for (int mt = 0; mt < 2; ++mt)
#pragma unroll
        for (int r = 0; r < 8; ++r) {
          const int m = mt * 16 + r + hi8;
          const float val = acc[mt][0][r] + bv;
          if (!isCenter) {
            const int dm = (int)((dompack >> ((mt * 8 + r) * 2)) & 3u);
            if (dm == gi) sSel[m * H3DIM + n] = val;     // h_domain (selected)
          } else {
            sSel[m * H3DIM + n] = val * tanhf(sSel[m * H3DIM + n]);  // fuse
          }
        }
    }
    __syncthreads();
  }

  // ---- final MLP 128->64->1 + aux net + sigmoid (32 rows x 8 threads) ----
  const int row = threadIdx.x >> 3;
  const int sub = threadIdx.x & 7;
  const float* fr = sSel + row * H3DIM;
  float partial = 0.0f;
#pragma unroll
  for (int jj = 0; jj < 8; ++jj) {
    const int j = sub + jj * 8;
    float acc = P.fb1[j];
    for (int n = 0; n < H3DIM; ++n) acc += fr[n] * P.fW1[n * 64 + j];
    acc = fmaxf(acc, 0.0f);
    partial += acc * P.fW2[j];
  }
#pragma unroll
  for (int m = 4; m >= 1; m >>= 1) partial += __shfl_xor(partial, m, 32);

  if (sub == 0) {
    float main_v = partial + P.fb2[0];
    const int d = P.dom[row0 + row];
    const float* de = P.dom_emb + d * 16;
    float aux = P.ab2[0];
    for (int j = 0; j < 32; ++j) {
      float a = P.ab1[j];
#pragma unroll
      for (int e = 0; e < 16; ++e) a += de[e] * P.aW1[e * 32 + j];
      aux += fmaxf(a, 0.0f) * P.aW2[j];
    }
    const float z = main_v + aux;
    P.out[row0 + row] = 1.0f / (1.0f + expf(-z));
  }
}

// ---------------------------------------------------------------------------
// Host launcher
// ---------------------------------------------------------------------------
extern "C" void kernel_launch(void* const* d_in, const int* in_sizes, int n_in,
                              void* d_out, int out_size, void* d_ws, size_t ws_size,
                              hipStream_t stream) {
  (void)in_sizes; (void)n_in; (void)out_size; (void)ws_size;

  const float* x      = (const float*)d_in[0];
  const int*   dom    = (const int*)  d_in[1];
  const float* pn_w   = (const float*)d_in[2];
  const float* pn_b   = (const float*)d_in[3];
  const float* cW1    = (const float*)d_in[4];
  const float* cb1    = (const float*)d_in[5];
  const float* cW2    = (const float*)d_in[6];
  const float* cb2    = (const float*)d_in[7];
  const float* cW3    = (const float*)d_in[8];
  const float* cb3    = (const float*)d_in[9];
  const float* dW1    = (const float*)d_in[10];
  const float* db1    = (const float*)d_in[11];
  const float* dW2    = (const float*)d_in[12];
  const float* db2    = (const float*)d_in[13];
  const float* dW3    = (const float*)d_in[14];
  const float* db3    = (const float*)d_in[15];
  const float* fW1    = (const float*)d_in[16];
  const float* fb1    = (const float*)d_in[17];
  const float* fW2    = (const float*)d_in[18];
  const float* fb2    = (const float*)d_in[19];
  const float* dom_emb= (const float*)d_in[20];
  const float* aW1    = (const float*)d_in[21];
  const float* ab1    = (const float*)d_in[22];
  const float* aW2    = (const float*)d_in[23];
  const float* ab2    = (const float*)d_in[24];

  _Float16* ws = (_Float16*)d_ws;
  _Float16* nrm  = ws + WS_NORM;
  _Float16* cW1t = ws + WS_CW1T;
  _Float16* cW2t = ws + WS_CW2T;
  _Float16* cW3t = ws + WS_CW3T;
  _Float16* dW1t = ws + WS_DW1T;
  _Float16* dW2t = ws + WS_DW2T;
  _Float16* dW3t = ws + WS_DW3T;

  // --- weight downconvert + transpose to (N,K) f16 ---
  {
    int t;
    t = DIN * H1DIM;
    transpose_to_f16<<<(t + 255) / 256, 256, 0, stream>>>(cW1, cW1t, DIN,   H1DIM, t);
    t = H1DIM * H2DIM;
    transpose_to_f16<<<(t + 255) / 256, 256, 0, stream>>>(cW2, cW2t, H1DIM, H2DIM, t);
    t = H2DIM * H3DIM;
    transpose_to_f16<<<(t + 255) / 256, 256, 0, stream>>>(cW3, cW3t, H2DIM, H3DIM, t);
    t = NDOM * DIN * H1DIM;
    transpose_to_f16<<<(t + 255) / 256, 256, 0, stream>>>(dW1, dW1t, DIN,   H1DIM, t);
    t = NDOM * H1DIM * H2DIM;
    transpose_to_f16<<<(t + 255) / 256, 256, 0, stream>>>(dW2, dW2t, H1DIM, H2DIM, t);
    t = NDOM * H2DIM * H3DIM;
    transpose_to_f16<<<(t + 255) / 256, 256, 0, stream>>>(dW3, dW3t, H2DIM, H3DIM, t);
  }

  // --- LayerNorm + partitioned affine -> f16 ---
  ln_affine_f16<<<BATCH / 8, 256, 0, stream>>>(x, dom, pn_w, pn_b, nrm);

  // --- fused WMMA chain ---
  Params P;
  P.nrm = nrm;
  P.cW1t = cW1t; P.cW2t = cW2t; P.cW3t = cW3t;
  P.dW1t = dW1t; P.dW2t = dW2t; P.dW3t = dW3t;
  P.dom = dom;
  P.cb1 = cb1; P.cb2 = cb2; P.cb3 = cb3;
  P.db1 = db1; P.db2 = db2; P.db3 = db3;
  P.fW1 = fW1; P.fb1 = fb1; P.fW2 = fW2; P.fb2 = fb2;
  P.dom_emb = dom_emb; P.aW1 = aW1; P.ab1 = ab1; P.aW2 = aW2; P.ab2 = ab2;
  P.out = (float*)d_out;

  fused_star<<<BATCH / MROWS, 256, 0, stream>>>(P);
}
